// SinglePointer_68118181314636
// MI455X (gfx1250) — matrix operand order repaired and checked by reading, not compiled, not run on hardware
//
#include <hip/hip_runtime.h>

// ---------------- problem constants ----------------
#define NOBJ 20
#define NFD  16
#define H    64
#define EPW  8                 // batch elements per workgroup
#define ROWS (EPW*NOBJ)        // 160 node-rows
#define TPB  320               // 10 wave32 -> one 16-row WMMA M-tile per wave
#define ASTR 136               // bf16 activation stride (padded vs 128 -> conflict free)
#define HSTR 68                // f32 h stride (padded vs 64)

typedef __attribute__((ext_vector_type(16))) __bf16       v16bf;
typedef __attribute__((ext_vector_type(8)))  float        v8f;
typedef __attribute__((ext_vector_type(4)))  unsigned int u32x4;

// ---------------- LDS layout (bytes) ----------------
#define OFF_HS   0
#define OFF_AS   (OFF_HS + ROWS*HSTR*4)        // 43520
#define OFF_MSG  (OFF_AS + ROWS*ASTR*2)        // 87040
#define OFF_G    (OFF_MSG + EPW*H*4)
#define OFF_Q    (OFF_G   + EPW*H*4)
#define OFF_WQ   (OFF_Q   + EPW*H*4)
#define OFF_WE   (OFF_WQ  + H*H*4)
#define OFF_WOP  (OFF_WE  + NFD*H*4)
#define OFF_WC1  (OFF_WOP + H*8*4)
#define OFF_WC2  (OFF_WC1 + H*10*4)
#define OFF_BIAS (OFF_WC2 + H*10*4)
#define N_BIASF  (64*11 + 8 + 10 + 10)         // 732 floats
#define OFF_RCP  (OFF_BIAS + N_BIASF*4)
#define OFF_CNT  (OFF_RCP + EPW*4)
#define SMEM_BYTES (OFF_CNT + EPW*4 + 16)      // ~123.8 KB -> 2 blocks/WGP

// swizzled bf16 weight offsets in workspace (halves)
#define WOFF_W1A 0
#define WOFF_W1B 8192
#define WOFF_W2A 12288
#define WOFF_W2B 20480
#define WOFF_WK  24576

static __device__ __forceinline__ unsigned short f2bf(float x){
  unsigned u = __builtin_bit_cast(unsigned, x);
  unsigned r = u + 0x7FFFu + ((u >> 16) & 1u);   // RNE
  return (unsigned short)(r >> 16);
}
static __device__ __forceinline__ float bf2f(unsigned short h){
  unsigned u = ((unsigned)h) << 16;
  return __builtin_bit_cast(float, u);
}

// A-operand fragment: halves [0..7] and [16..23] relative to p (caller adds base8)
static __device__ __forceinline__ v16bf load_a16(const unsigned short* p){
  union { u32x4 u[2]; v16bf v; } x;
  x.u[0] = *(const u32x4*)(p);
  x.u[1] = *(const u32x4*)(p + 16);
  return x.v;
}
// B-operand fragment: 32 contiguous bytes per lane from the swizzled weight image
static __device__ __forceinline__ v16bf load_b16(const unsigned short* __restrict__ w,
                                                 int frag, int lane){
  const u32x4* p = (const u32x4*)(w + (((size_t)frag * 32 + lane) << 4));
  union { u32x4 u[2]; v16bf v; } x;
  x.u[0] = p[0]; x.u[1] = p[1];
  return x.v;
}
static __device__ __forceinline__ v8f wmma_bf16(v16bf a, v16bf b, v8f c){
  return __builtin_amdgcn_wmma_f32_16x16x32_bf16(false, a, false, b, (short)0, c, false, false);
}

// ---------------- prep: f32 [K][N] -> bf16 swizzled B-operand fragments ----------------
// frag f = kt*(N/16)+ntj ; lane l ; half i : K = kt*32 + 8*(l>>4) + (i&7) + 16*(i>>3), Ncol = ntj*16 + (l&15)
__global__ void swizzle_w(const float* __restrict__ src, unsigned short* __restrict__ dst,
                          int K, int Ncols){
  int nt = Ncols >> 4;
  int total = (K >> 5) * nt * 512;
  for (int t = blockIdx.x * blockDim.x + threadIdx.x; t < total; t += gridDim.x * blockDim.x){
    int i = t & 15, l = (t >> 4) & 31, f = t >> 9;
    int ntj = f % nt, kti = f / nt;
    int k = kti * 32 + ((l >> 4) << 3) + (i & 7) + ((i >> 3) << 4);
    int n = ntj * 16 + (l & 15);
    dst[t] = f2bf(src[(size_t)k * Ncols + n]);
  }
}

// ---------------- fused GNN kernel ----------------
__global__ __launch_bounds__(TPB, 1)
void gnn_fused(const float* __restrict__ nf, const float* __restrict__ W_embed,
               const float* __restrict__ b_embed,
               const float* __restrict__ b1a, const float* __restrict__ b1b,
               const float* __restrict__ g1,  const float* __restrict__ be1,
               const float* __restrict__ b2a, const float* __restrict__ b2b,
               const float* __restrict__ g2,  const float* __restrict__ be2,
               const float* __restrict__ Wop, const float* __restrict__ bop,
               const float* __restrict__ Wc1, const float* __restrict__ bc1,
               const float* __restrict__ Wc2, const float* __restrict__ bc2,
               const float* __restrict__ Wq,  const float* __restrict__ bq,
               const float* __restrict__ bk,  const int* __restrict__ nn_c,
               const unsigned short* __restrict__ wW1a, const unsigned short* __restrict__ wW1b,
               const unsigned short* __restrict__ wW2a, const unsigned short* __restrict__ wW2b,
               const unsigned short* __restrict__ wWk,
               float* __restrict__ out_op, float* __restrict__ out_c1,
               float* __restrict__ out_c2, float* __restrict__ out_ptr)
{
  extern __shared__ unsigned char smem[];
  float*          hS   = (float*)(smem + OFF_HS);
  unsigned short* aS   = (unsigned short*)(smem + OFF_AS);
  float*          msgS = (float*)(smem + OFF_MSG);
  float*          gS   = (float*)(smem + OFF_G);
  float*          qS   = (float*)(smem + OFF_Q);
  float*          WqS  = (float*)(smem + OFF_WQ);
  float*          WeS  = (float*)(smem + OFF_WE);
  float*          WopS = (float*)(smem + OFF_WOP);
  float*          Wc1S = (float*)(smem + OFF_WC1);
  float*          Wc2S = (float*)(smem + OFF_WC2);
  float*          bB   = (float*)(smem + OFF_BIAS);
  float*          rcpS = (float*)(smem + OFF_RCP);
  int*            cntS = (int*)(smem + OFF_CNT);

  float* bEmbS = bB;        float* b1aS = bB + 64;  float* b1bS = bB + 128;
  float* g1S   = bB + 192;  float* be1S = bB + 256; float* b2aS = bB + 320;
  float* b2bS  = bB + 384;  float* g2S  = bB + 448; float* be2S = bB + 512;
  float* bqS   = bB + 576;  float* bkS  = bB + 640; float* bopS = bB + 704;
  float* bc1S  = bB + 712;  float* bc2S = bB + 722;

  const int tid = threadIdx.x;
  const int e0  = blockIdx.x * EPW;

  // ---- stage small f32 weights / biases / counts ----
  for (int i = tid; i < H*H;    i += TPB) WqS[i]  = Wq[i];
  for (int i = tid; i < NFD*H;  i += TPB) WeS[i]  = W_embed[i];
  for (int i = tid; i < H*8;    i += TPB) WopS[i] = Wop[i];
  for (int i = tid; i < H*10;   i += TPB) Wc1S[i] = Wc1[i];
  for (int i = tid; i < H*10;   i += TPB) Wc2S[i] = Wc2[i];
  if (tid < 64){
    bEmbS[tid]=b_embed[tid]; b1aS[tid]=b1a[tid]; b1bS[tid]=b1b[tid];
    g1S[tid]=g1[tid]; be1S[tid]=be1[tid]; b2aS[tid]=b2a[tid]; b2bS[tid]=b2b[tid];
    g2S[tid]=g2[tid]; be2S[tid]=be2[tid]; bqS[tid]=bq[tid];  bkS[tid]=bk[tid];
  }
  if (tid < 8)  bopS[tid] = bop[tid];
  if (tid < 10){ bc1S[tid] = bc1[tid]; bc2S[tid] = bc2[tid]; }
  if (tid < EPW){
    int c = nn_c[e0 + tid];
    cntS[tid] = c;
    rcpS[tid] = 1.0f / fmaxf((float)c, 1.0f);
  }

  // ---- embed: h = nf @ W_embed + b (K=16 -> VALU); stage nf into aS scratch ----
  float* nfS = (float*)aS;                       // alias, overwritten later
  const float* nfg = nf + (size_t)e0 * NOBJ * NFD;
  for (int i = tid; i < ROWS*NFD; i += TPB) nfS[i] = nfg[i];
  __syncthreads();
  for (int idx = tid; idx < ROWS*H; idx += TPB){
    int r = idx >> 6, c = idx & 63;
    float acc = bEmbS[c];
    const float* row = nfS + r * NFD;
    #pragma unroll
    for (int k = 0; k < NFD; ++k) acc += row[k] * WeS[k*H + c];
    hS[r*HSTR + c] = acc;
  }
  __syncthreads();

  const int wv   = tid >> 5;
  const int lane = tid & 31;
  const int mrow = wv * 16;
  const int hi8  = (lane >> 4) << 3;                 // 0 | 8
  const unsigned short* arow = aS + (size_t)(mrow + (lane & 15)) * ASTR + hi8;

  // ================= two message-passing rounds =================
  for (int rd = 0; rd < 2; ++rd){
    const unsigned short* wA = rd ? wW2a : wW1a;
    const unsigned short* wB = rd ? wW2b : wW1b;
    const float* baS  = rd ? b2aS : b1aS;
    const float* bbS  = rd ? b2bS : b1bS;
    const float* ggS  = rd ? g2S  : g1S;
    const float* beS  = rd ? be2S : be1S;

    // masked-mean message
    for (int idx = tid; idx < EPW*H; idx += TPB){
      int e = idx >> 6, c = idx & 63, cn = cntS[e];
      float s = 0.f;
      for (int n = 0; n < NOBJ; ++n) if (n < cn) s += hS[(e*NOBJ+n)*HSTR + c];
      msgS[idx] = s * rcpS[e];
    }
    __syncthreads();
    // build x = [h | msg] in bf16
    for (int idx = tid; idx < ROWS*H; idx += TPB){
      int r = idx >> 6, c = idx & 63;
      aS[r*ASTR + c]      = f2bf(hS[r*HSTR + c]);
      aS[r*ASTR + 64 + c] = f2bf(msgS[(r/NOBJ)*H + c]);
    }
    __syncthreads();

    // GEMM1: t = relu(x @ Wa + ba), [16,128]@[128,64] per wave tile
    v16bf a0 = load_a16(arow);
    v16bf a1 = load_a16(arow + 32);
    v16bf a2 = load_a16(arow + 64);
    v16bf a3 = load_a16(arow + 96);
    #pragma unroll
    for (int nt2 = 0; nt2 < 4; ++nt2){
      v8f acc = {0.f,0.f,0.f,0.f,0.f,0.f,0.f,0.f};
      acc = wmma_bf16(a0, load_b16(wA, 0*4 + nt2, lane), acc);
      acc = wmma_bf16(a1, load_b16(wA, 1*4 + nt2, lane), acc);
      acc = wmma_bf16(a2, load_b16(wA, 2*4 + nt2, lane), acc);
      acc = wmma_bf16(a3, load_b16(wA, 3*4 + nt2, lane), acc);
      const int   col = nt2*16 + (lane & 15);
      const float bav = baS[col];
      #pragma unroll
      for (int v = 0; v < 8; ++v){
        float t = acc[v] + bav;  t = t > 0.f ? t : 0.f;
        aS[(size_t)(mrow + v + hi8)*ASTR + col] = f2bf(t);   // t over x's h half
      }
    }
    // GEMM2: h += t @ Wb + bb
    v16bf t0 = load_a16(arow);
    v16bf t1 = load_a16(arow + 32);
    #pragma unroll
    for (int nt2 = 0; nt2 < 4; ++nt2){
      v8f acc = {0.f,0.f,0.f,0.f,0.f,0.f,0.f,0.f};
      acc = wmma_bf16(t0, load_b16(wB, 0*4 + nt2, lane), acc);
      acc = wmma_bf16(t1, load_b16(wB, 1*4 + nt2, lane), acc);
      const int   col = nt2*16 + (lane & 15);
      const float bbv = bbS[col];
      #pragma unroll
      for (int v = 0; v < 8; ++v)
        hS[(size_t)(mrow + v + hi8)*HSTR + col] += acc[v] + bbv;
    }
    __syncthreads();

    // LayerNorm + mask, one row per thread
    if (tid < ROWS){
      float* row = hS + tid*HSTR;
      float mu = 0.f;
      for (int c = 0; c < H; ++c) mu += row[c];
      mu *= (1.f/H);
      float var = 0.f;
      for (int c = 0; c < H; ++c){ float d = row[c]-mu; var += d*d; }
      var *= (1.f/H);
      float rs = rsqrtf(var + 1e-5f);
      int e = tid / NOBJ, n = tid % NOBJ;
      float m = (n < cntS[e]) ? 1.f : 0.f;
      for (int c = 0; c < H; ++c)
        row[c] = ((row[c]-mu)*rs*ggS[c] + beS[c]) * m;
    }
    __syncthreads();
  }

  // ================= readout =================
  for (int idx = tid; idx < EPW*H; idx += TPB){    // g (rows already masked)
    int e = idx >> 6, c = idx & 63;
    float s = 0.f;
    for (int n = 0; n < NOBJ; ++n) s += hS[(e*NOBJ+n)*HSTR + c];
    gS[idx] = s * rcpS[e];
  }
  __syncthreads();
  for (int idx = tid; idx < EPW*H; idx += TPB){    // q = g@Wq + bq
    int e = idx >> 6, o = idx & 63;
    float s = bqS[o];
    const float* gv = gS + e*H;
    for (int j = 0; j < H; ++j) s += gv[j] * WqS[j*H + o];
    qS[idx] = s;
  }
  for (int idx = tid; idx < EPW*28; idx += TPB){   // heads
    int e = idx / 28, o = idx % 28;
    const float* gv = gS + e*H;
    if (o < 8){
      float s = bopS[o];
      for (int j = 0; j < H; ++j) s += gv[j] * WopS[j*8 + o];
      out_op[(size_t)(e0+e)*8 + o] = s;
    } else if (o < 18){
      int oo = o - 8; float s = bc1S[oo];
      for (int j = 0; j < H; ++j) s += gv[j] * Wc1S[j*10 + oo];
      out_c1[(size_t)(e0+e)*10 + oo] = s;
    } else {
      int oo = o - 18; float s = bc2S[oo];
      for (int j = 0; j < H; ++j) s += gv[j] * Wc2S[j*10 + oo];
      out_c2[(size_t)(e0+e)*10 + oo] = s;
    }
  }
  for (int idx = tid; idx < ROWS*H; idx += TPB){   // rebuild bf16 h for Wk GEMM
    int r = idx >> 6, c = idx & 63;
    aS[r*ASTR + c] = f2bf(hS[r*HSTR + c]);
  }
  __syncthreads();

  // k = h @ Wk + bk (WMMA), store bf16 into aS cols 64..127
  {
    v16bf t0 = load_a16(arow);
    v16bf t1 = load_a16(arow + 32);
    #pragma unroll
    for (int nt2 = 0; nt2 < 4; ++nt2){
      v8f acc = {0.f,0.f,0.f,0.f,0.f,0.f,0.f,0.f};
      acc = wmma_bf16(t0, load_b16(wWk, 0*4 + nt2, lane), acc);
      acc = wmma_bf16(t1, load_b16(wWk, 1*4 + nt2, lane), acc);
      const int   col = nt2*16 + (lane & 15);
      const float bkv = bkS[col];
      #pragma unroll
      for (int v = 0; v < 8; ++v)
        aS[(size_t)(mrow + v + hi8)*ASTR + 64 + col] = f2bf(acc[v] + bkv);
    }
  }
  __syncthreads();

  // ptr[b,n] = q(e) . k(row), masked
  if (tid < ROWS){
    int e = tid / NOBJ, n = tid % NOBJ;
    const unsigned short* kr = aS + (size_t)tid*ASTR + 64;
    const float* qv = qS + e*H;
    float s = 0.f;
    for (int c = 0; c < H; ++c) s += qv[c] * bf2f(kr[c]);
    out_ptr[(size_t)(e0+e)*NOBJ + n] = (n < cntS[e]) ? s : -1e9f;
  }
}

// ---------------- host entry ----------------
extern "C" void kernel_launch(void* const* d_in, const int* in_sizes, int n_in,
                              void* d_out, int out_size, void* d_ws, size_t ws_size,
                              hipStream_t stream) {
  const float* nf      = (const float*)d_in[0];
  const float* W_embed = (const float*)d_in[1];
  const float* b_embed = (const float*)d_in[2];
  const float* W1a = (const float*)d_in[3];  const float* b1a = (const float*)d_in[4];
  const float* W1b = (const float*)d_in[5];  const float* b1b = (const float*)d_in[6];
  const float* g1  = (const float*)d_in[7];  const float* be1 = (const float*)d_in[8];
  const float* W2a = (const float*)d_in[9];  const float* b2a = (const float*)d_in[10];
  const float* W2b = (const float*)d_in[11]; const float* b2b = (const float*)d_in[12];
  const float* g2  = (const float*)d_in[13]; const float* be2 = (const float*)d_in[14];
  const float* Wop = (const float*)d_in[15]; const float* bop = (const float*)d_in[16];
  const float* Wc1 = (const float*)d_in[17]; const float* bc1 = (const float*)d_in[18];
  const float* Wc2 = (const float*)d_in[19]; const float* bc2 = (const float*)d_in[20];
  const float* Wq  = (const float*)d_in[21]; const float* bq  = (const float*)d_in[22];
  const float* Wk  = (const float*)d_in[23]; const float* bk  = (const float*)d_in[24];
  const int*  nn_c = (const int*)d_in[25];

  const int nb = in_sizes[25];                      // batch size
  unsigned short* wbf = (unsigned short*)d_ws;      // swizzled bf16 weights (~56 KB)

  swizzle_w<<<32, 256, 0, stream>>>(W1a, wbf + WOFF_W1A, 128, 64);
  swizzle_w<<<16, 256, 0, stream>>>(W1b, wbf + WOFF_W1B,  64, 64);
  swizzle_w<<<32, 256, 0, stream>>>(W2a, wbf + WOFF_W2A, 128, 64);
  swizzle_w<<<16, 256, 0, stream>>>(W2b, wbf + WOFF_W2B,  64, 64);
  swizzle_w<<<16, 256, 0, stream>>>(Wk,  wbf + WOFF_WK,   64, 64);

  float* out = (float*)d_out;
  gnn_fused<<<nb / EPW, TPB, SMEM_BYTES, stream>>>(
      nf, W_embed, b_embed, b1a, b1b, g1, be1, b2a, b2b, g2, be2,
      Wop, bop, Wc1, bc1, Wc2, bc2, Wq, bq, bk, nn_c,
      wbf + WOFF_W1A, wbf + WOFF_W1B, wbf + WOFF_W2A, wbf + WOFF_W2B, wbf + WOFF_WK,
      out, out + (size_t)nb*8, out + (size_t)nb*18, out + (size_t)nb*28);
}
